// CnnLstmCrf_83923660963860
// MI455X (gfx1250) — matrix operand order repaired and checked by reading, not compiled, not run on hardware
//
#include <hip/hip_runtime.h>
#include <hip/hip_bf16.h>

typedef __attribute__((ext_vector_type(16))) _Float16 v16h;
typedef __attribute__((ext_vector_type(8)))  _Float16 v8h;
typedef __attribute__((ext_vector_type(8)))  float    v8f;

// Problem constants
constexpr int Bc   = 64;
constexpr int Sc   = 250;
constexpr int CLc  = 16;
constexpr int CDc  = 300;
constexpr int CHc  = 50;
constexpr int WDc  = 300;
constexpr int FDc  = 5;
constexpr int INc  = 355;     // WD + CH + FD
constexpr int KXP  = 384;     // IN padded to 32
constexpr int Hc   = 100;
constexpr int KHP  = 128;     // H padded
constexpr int NG   = 400;     // 4*H gate columns (interleaved i,f,g,o per unit)
constexpr int Tc   = 22;
constexpr int KC   = 928;     // 3*CD padded to 32
constexpr int NW   = Bc * Sc; // 16000 words

__device__ __forceinline__ v8f wmma_f16(v16h a, v16h b, v8f c) {
  return __builtin_amdgcn_wmma_f32_16x16x32_f16(false, a, false, b, (short)0, c, false, false);
}

// A fragment, row-major [M][K]. Lane group0: K kb..kb+7, kb+16..kb+23 ; group1: +8.
__device__ __forceinline__ v16h frag_a(const _Float16* A, int lda, int mbase, int kbase) {
  int lane = threadIdx.x & 31;
  const _Float16* p = A + (size_t)(mbase + (lane & 15)) * lda + kbase + ((lane >> 4) << 3);
  v8h lo = *(const v8h*)p;
  v8h hi = *(const v8h*)(p + 16);
  return __builtin_shufflevector(lo, hi, 0,1,2,3,4,5,6,7,8,9,10,11,12,13,14,15);
}

// B fragment from TRANSPOSED weights Bt[N][K]. Lane group0: K kb..kb+7, kb+16..kb+23 ; group1: +8.
__device__ __forceinline__ v16h frag_bt(const _Float16* Bt, int ldk, int nbase, int kbase) {
  int lane = threadIdx.x & 31;
  const _Float16* p = Bt + (size_t)(nbase + (lane & 15)) * ldk + kbase + ((lane >> 4) << 3);
  v8h lo = *(const v8h*)p;
  v8h hi = *(const v8h*)(p + 16);
  return __builtin_shufflevector(lo, hi, 0,1,2,3,4,5,6,7,8,9,10,11,12,13,14,15);
}

// ---------------- weight preprocessing (f32 -> f16, transpose, gate interleave) -------------
__global__ void k_prep(const float* conv_w, const float* wihf, const float* wihb,
                       const float* whhf, const float* whhb, const float* bf, const float* bb2,
                       const float* projw,
                       _Float16* Wc, _Float16* Wih, _Float16* Whh, _Float16* Wproj, float* biasil) {
  int tid = blockIdx.x * blockDim.x + threadIdx.x;
  int nth = gridDim.x * blockDim.x;
  // Wc[o 0..63][kk 0..927], kk = k*CD + d  (im2col B, transposed layout)
  for (int i = tid; i < 64 * KC; i += nth) {
    int o = i / KC, kk = i % KC;
    float v = 0.f;
    if (o < CHc && kk < 3 * CDc) { int k = kk / CDc, d = kk % CDc; v = conv_w[(o * CDc + d) * 3 + k]; }
    Wc[i] = (_Float16)v;
  }
  // Wih[dir*NG + n][k 0..383]; n interleaved: n=4u+q -> orig row r=q*H+u
  for (int i = tid; i < 2 * NG * KXP; i += nth) {
    int n2 = i / KXP, k = i % KXP;
    int dir = n2 / NG, n = n2 % NG;
    int r = (n & 3) * Hc + (n >> 2);
    const float* w = dir ? wihb : wihf;
    Wih[i] = (_Float16)((k < INc) ? w[r * INc + k] : 0.f);
  }
  // Whh[dir][n 0..399][k 0..127]
  for (int i = tid; i < 2 * NG * KHP; i += nth) {
    int n2 = i / KHP, k = i % KHP;
    int dir = n2 / NG, n = n2 % NG;
    int r = (n & 3) * Hc + (n >> 2);
    const float* w = dir ? whhb : whhf;
    Whh[i] = (_Float16)((k < Hc) ? w[r * Hc + k] : 0.f);
  }
  // Wproj[n 0..31][k 0..255]: k<100 -> projw row k ; 128<=k<228 -> projw row k-28
  for (int i = tid; i < 32 * 256; i += nth) {
    int n = i / 256, k = i % 256;
    float v = 0.f;
    if (n < Tc) {
      if (k < Hc) v = projw[k * Tc + n];
      else if (k >= KHP && k < KHP + Hc) v = projw[(k - 28) * Tc + n];
    }
    Wproj[i] = (_Float16)v;
  }
  // interleaved bias [800]
  for (int i = tid; i < 2 * NG; i += nth) {
    int dir = i / NG, n = i % NG;
    int r = (n & 3) * Hc + (n >> 2);
    biasil[i] = dir ? bb2[r] : bf[r];
  }
}

// ---------------- char CNN as im2col GEMM + max-pool: one word per block -------------------
__global__ __launch_bounds__(128) void k_charconv(const int* batch_char, const int* recover,
                                                  const float* char_emb, const float* conv_b,
                                                  const _Float16* Wc, _Float16* Ax) {
  __shared__ __align__(16) _Float16 Aim[16 * KC];
  __shared__ int ch[16];
  int w = blockIdx.x;
  int tid = threadIdx.x;
  if (tid < 16) {
    int rec = recover[w];
    ch[tid] = batch_char[rec * CLc + tid];
  }
  __syncthreads();
  for (int i = tid; i < 16 * KC; i += 128) {
    int p = i / KC, kk = i % KC;
    float v = 0.f;
    if (kk < 3 * CDc) {
      int k = kk / CDc, d = kk % CDc;
      int q = p + k - 1;
      if (q >= 0 && q < CLc) v = char_emb[(size_t)ch[q] * CDc + d];
    }
    Aim[i] = (_Float16)v;
  }
  __syncthreads();
  int wv = tid >> 5, lane = tid & 31;
  v8f acc = {};
  for (int ks = 0; ks < KC / 32; ++ks) {
    v16h a = frag_a(Aim, KC, 0, ks * 32);
    v16h b = frag_bt(Wc, KC, wv * 16, ks * 32);
    acc = wmma_f16(a, b, acc);
  }
  float m = acc[0];
#pragma unroll
  for (int r = 1; r < 8; ++r) m = fmaxf(m, acc[r]);
  m = fmaxf(m, __shfl_xor(m, 16));
  int o = wv * 16 + (lane & 15);
  if (lane < 16 && o < CHc)
    Ax[(size_t)w * KXP + WDc + o] = (_Float16)(m + conv_b[o]);
}

// ---------------- embedding gathers into f16 A matrix (cols 300..349 from char CNN) --------
__global__ void k_gather(const int* bw, const int* bfeat, const float* wemb, const float* femb,
                         _Float16* Ax) {
  int tid = blockIdx.x * blockDim.x + threadIdx.x;
  int nth = gridDim.x * blockDim.x;
  for (long long i = tid; i < (long long)NW * KXP; i += nth) {
    int w = (int)(i / KXP), k = (int)(i % KXP);
    if (k < WDc)                  Ax[i] = (_Float16)wemb[(size_t)bw[w] * WDc + k];
    else if (k >= WDc + CHc && k < INc) Ax[i] = (_Float16)femb[bfeat[w] * FDc + (k - WDc - CHc)];
    else if (k >= INc)            Ax[i] = (_Float16)0.f;
    // 300..349 written by k_charconv
  }
}

// ------- gates_x = word_rep @ W_ih^T + b, stored time-major [t][b][800] for the scan -------
__global__ __launch_bounds__(32) void k_gates(const _Float16* Ax, const _Float16* Wih,
                                              const float* biasil, float* gates) {
  int mb = blockIdx.x * 16, nb = blockIdx.y * 16;
  v8f acc = {};
  for (int ks = 0; ks < KXP / 32; ++ks) {
    v16h a = frag_a(Ax, KXP, mb, ks * 32);
    v16h b = frag_bt(Wih, KXP, nb, ks * 32);
    acc = wmma_f16(a, b, acc);
  }
  int lane = threadIdx.x & 31;
  int n = nb + (lane & 15), hi = lane >> 4;
  float bias = biasil[n];
#pragma unroll
  for (int r = 0; r < 8; ++r) {
    int w = mb + r + hi * 8;            // word index = b*S + t
    int b = w / Sc, t = w % Sc;
    gates[((size_t)t * Bc + b) * 800 + n] = acc[r] + bias;   // time-major
  }
}

// ---------------- persistent BiLSTM recurrence: 1 workgroup per direction ------------------
__global__ __launch_bounds__(1024) void k_lstm(const float* gates, const _Float16* Whh,
                                               const int* lens, _Float16* Aproj) {
  __shared__ __align__(16) _Float16 hbuf[2][Bc * KHP];   // 32 KB double-buffered h (f16)
  __shared__ __align__(16) float scratch[32][256];        // 32 KB per-wave gate tiles
  int dir = blockIdx.x;
  int tid = threadIdx.x, wv = tid >> 5, lane = tid & 31;
  const _Float16* Wd = Whh + (size_t)dir * NG * KHP;

  for (int i = tid; i < 2 * Bc * KHP; i += 1024) (&hbuf[0][0])[i] = (_Float16)0.f;

  float creg[4][2], hreg[4][2];
  int lenv[4][2];
#pragma unroll
  for (int i = 0; i < 4; ++i) {
    int tl = wv + i * 32;
    if (tl < 100) {
#pragma unroll
      for (int ps = 0; ps < 2; ++ps) {
        int p = lane + ps * 32;
        int bb = (tl / 25) * 16 + (p >> 2);
        lenv[i][ps] = lens[bb];
        creg[i][ps] = 0.f; hreg[i][ps] = 0.f;
      }
    }
  }
  __syncthreads();

  for (int step = 0; step < Sc; ++step) {
    int t = dir ? (Sc - 1 - step) : step;
    int cur = step & 1, nxt = cur ^ 1;

    // prefetch next timestep's contiguous 204.8 KB gate block (overlap with WMMA below)
    int tn = dir ? (t - 1) : (t + 1);
    if (tn >= 0 && tn < Sc) {
      const char* pf = (const char*)(gates + (size_t)tn * Bc * 800);
      __builtin_prefetch(pf + (size_t)tid * 200, 0, 1);
    }

    for (int i = 0; i < 4; ++i) {
      int tl = wv + i * 32;
      if (tl >= 100) break;
      int mt = tl / 25, nt = tl % 25;
      v8f acc = {};
#pragma unroll
      for (int ks = 0; ks < KHP / 32; ++ks) {
        v16h a = frag_a(hbuf[cur], KHP, mt * 16, ks * 32);  // LDS
        v16h b = frag_bt(Wd, KHP, nt * 16, ks * 32);        // L2-resident
        acc = wmma_f16(a, b, acc);
      }
      int n = lane & 15, hi = lane >> 4;
      int col = dir * NG + nt * 16 + n;
#pragma unroll
      for (int r = 0; r < 8; ++r) {
        int bb = mt * 16 + r + hi * 8;
        float g = acc[r] + gates[((size_t)t * Bc + bb) * 800 + col];  // time-major read
        scratch[wv][(r + hi * 8) * 16 + n] = g;
      }
      asm volatile("s_wait_dscnt 0" ::: "memory");   // own-wave LDS RAW fence
#pragma unroll
      for (int ps = 0; ps < 2; ++ps) {
        int p = lane + ps * 32;
        int ml = p >> 2, ul = p & 3;
        const float* g4 = &scratch[wv][ml * 16 + ul * 4];  // i,f,g,o for this unit
        float gi = g4[0], gf = g4[1], gg = g4[2], go = g4[3];
        float co = creg[i][ps], ho = hreg[i][ps];
        float si = 1.f / (1.f + __expf(-gi));
        float sf = 1.f / (1.f + __expf(-gf));
        float so = 1.f / (1.f + __expf(-go));
        float cn = sf * co + si * tanhf(gg);
        float hn = so * tanhf(cn);
        bool valid = t < lenv[i][ps];
        float cw = valid ? cn : co, hw = valid ? hn : ho;
        creg[i][ps] = cw; hreg[i][ps] = hw;
        int bb = mt * 16 + ml;
        int ug = nt * 4 + ul;
        hbuf[nxt][bb * KHP + ug] = (_Float16)hw;
        Aproj[((size_t)bb * Sc + t) * 256 + dir * KHP + ug] = (_Float16)hw;
      }
    }
    __syncthreads();   // one barrier per timestep
  }
}

// ---------------- feats = lstm_out @ proj_w + proj_b --------------------------------------
__global__ __launch_bounds__(32) void k_proj(const _Float16* Aproj, const _Float16* Wproj,
                                             const float* projb, float* feats) {
  int mb = blockIdx.x * 16, nb = blockIdx.y * 16;
  v8f acc = {};
  for (int ks = 0; ks < 256 / 32; ++ks) {
    v16h a = frag_a(Aproj, 256, mb, ks * 32);
    v16h b = frag_bt(Wproj, 256, nb, ks * 32);
    acc = wmma_f16(a, b, acc);
  }
  int lane = threadIdx.x & 31;
  int n = nb + (lane & 15), hi = lane >> 4;
  float bias = (n < Tc) ? projb[n] : 0.f;
#pragma unroll
  for (int r = 0; r < 8; ++r)
    feats[(size_t)(mb + r + hi * 8) * 32 + n] = acc[r] + bias;
}

// ---------------- Viterbi decode: one wave per batch element ------------------------------
__global__ __launch_bounds__(32) void k_viterbi(const float* feats, const float* trans,
                                                const int* lens, int* bp, float* out) {
  __shared__ float tr[Tc * Tc];
  __shared__ float delta[32];
  __shared__ float fin[32];
  int b = blockIdx.x, lane = threadIdx.x;
  for (int i = lane; i < Tc * Tc; i += 32) tr[i] = trans[i];
  int len = lens[b];
  __syncthreads();
  if (lane < Tc) delta[lane] = tr[20 * Tc + lane] + feats[(size_t)b * Sc * 32 + lane];  // START=20
  __syncthreads();
  for (int t = 1; t < Sc; ++t) {
    float best = -3.4e38f; int arg = 0;
    if (lane < Tc) {
      for (int i2 = 0; i2 < Tc; ++i2) {
        float v = delta[i2] + tr[i2 * Tc + lane];
        if (v > best) { best = v; arg = i2; }     // first-max like jnp.argmax
      }
    }
    __syncthreads();
    if (lane < Tc) {
      bool valid = t < len;
      float nd = valid ? (best + feats[((size_t)b * Sc + t) * 32 + lane]) : delta[lane];
      bp[((size_t)b * (Sc - 1) + (t - 1)) * Tc + lane] = valid ? arg : lane;
      delta[lane] = nd;
    }
    __syncthreads();
  }
  if (lane < Tc) fin[lane] = delta[lane] + tr[lane * Tc + 21];  // STOP=21
  __syncthreads();
  if (lane == 0) {
    float best = -3.4e38f; int tag = 0;
    for (int j = 0; j < Tc; ++j) if (fin[j] > best) { best = fin[j]; tag = j; }
    out[(size_t)b * Sc + (Sc - 1)] = (float)tag;
    for (int t = Sc - 2; t >= 0; --t) {
      tag = bp[((size_t)b * (Sc - 1) + t) * Tc + tag];
      out[(size_t)b * Sc + t] = (float)tag;
    }
  }
}

extern "C" void kernel_launch(void* const* d_in, const int* in_sizes, int n_in,
                              void* d_out, int out_size, void* d_ws, size_t ws_size,
                              hipStream_t stream) {
  (void)in_sizes; (void)n_in; (void)out_size; (void)ws_size;
  const int*   batch_word = (const int*)d_in[0];
  const int*   batch_feat = (const int*)d_in[1];
  const int*   lens       = (const int*)d_in[2];
  const int*   batch_char = (const int*)d_in[3];
  const int*   recover    = (const int*)d_in[5];
  const float* char_emb   = (const float*)d_in[7];
  const float* word_emb   = (const float*)d_in[8];
  const float* feat_emb   = (const float*)d_in[9];
  const float* conv_w     = (const float*)d_in[10];
  const float* conv_b     = (const float*)d_in[11];
  const float* w_ih_f     = (const float*)d_in[12];
  const float* w_hh_f     = (const float*)d_in[13];
  const float* b_f        = (const float*)d_in[14];
  const float* w_ih_b     = (const float*)d_in[15];
  const float* w_hh_b     = (const float*)d_in[16];
  const float* b_b        = (const float*)d_in[17];
  const float* proj_w     = (const float*)d_in[18];
  const float* proj_b     = (const float*)d_in[19];
  const float* trans      = (const float*)d_in[20];

  char* ws = (char*)d_ws;
  size_t off = 0;
  auto alloc = [&](size_t bytes) { void* p = ws + off; off = (off + bytes + 255) & ~(size_t)255; return p; };
  _Float16* Wc     = (_Float16*)alloc((size_t)64 * KC * 2);
  _Float16* Wih    = (_Float16*)alloc((size_t)2 * NG * KXP * 2);
  _Float16* Whh    = (_Float16*)alloc((size_t)2 * NG * KHP * 2);
  _Float16* Wproj  = (_Float16*)alloc((size_t)32 * 256 * 2);
  float*    biasil = (float*)   alloc((size_t)2 * NG * 4);
  _Float16* Ax     = (_Float16*)alloc((size_t)NW * KXP * 2);
  float*    gates  = (float*)   alloc((size_t)NW * 800 * 4);
  _Float16* Aproj  = (_Float16*)alloc((size_t)NW * 256 * 2);
  float*    feats  = (float*)   alloc((size_t)NW * 32 * 4);
  int*      bp     = (int*)     alloc((size_t)Bc * (Sc - 1) * Tc * 4);

  k_prep<<<256, 256, 0, stream>>>(conv_w, w_ih_f, w_ih_b, w_hh_f, w_hh_b, b_f, b_b, proj_w,
                                  Wc, Wih, Whh, Wproj, biasil);
  k_gather<<<4096, 256, 0, stream>>>(batch_word, batch_feat, word_emb, feat_emb, Ax);
  k_charconv<<<NW, 128, 0, stream>>>(batch_char, recover, char_emb, conv_b, Wc, Ax);
  k_gates<<<dim3(NW / 16, 800 / 16), 32, 0, stream>>>(Ax, Wih, biasil, gates);
  hipMemsetAsync(Aproj, 0, (size_t)NW * 256 * 2, stream);   // zero pad cols 100..127 / 228..255
  k_lstm<<<2, 1024, 0, stream>>>(gates, Whh, lens, Aproj);
  k_proj<<<dim3(NW / 16, 2), 32, 0, stream>>>(Aproj, Wproj, proj_b, feats);
  k_viterbi<<<Bc, 32, 0, stream>>>(feats, trans, lens, bp, (float*)d_out);
}
